// Orthonorm_17781164605753
// MI455X (gfx1250) — matrix operand order repaired and checked by reading, not compiled, not run on hardware
//
#include <hip/hip_runtime.h>
#include <math.h>

#define UNITS 512
#define EPS_JITTER 1e-6f
#define SQRT_UNITS 22.62741699796952f

typedef __bf16 v16bf __attribute__((ext_vector_type(16)));
typedef __bf16 v8bf  __attribute__((ext_vector_type(8)));
typedef __bf16 v2bf  __attribute__((ext_vector_type(2)));
typedef float  v8f   __attribute__((ext_vector_type(8)));

__device__ __forceinline__ v16bf cat16(v8bf lo, v8bf hi) {
  return __builtin_shufflevector(lo, hi, 0, 1, 2, 3, 4, 5, 6, 7,
                                 8, 9, 10, 11, 12, 13, 14, 15);
}

// ---------------------------------------------------------------------------
// init: G = eps * I
// ---------------------------------------------------------------------------
__global__ void init_g_kernel(float* __restrict__ G) {
  int idx = blockIdx.x * blockDim.x + threadIdx.x;
  if (idx < UNITS * UNITS) {
    int i = idx >> 9, j = idx & 511;
    G[idx] = (i == j) ? EPS_JITTER : 0.0f;
  }
}

// ---------------------------------------------------------------------------
// Gram: lower-triangle of G = x^T x, bf16 WMMA, split-K with f32 atomics.
// Only blocks bi >= bj are computed (downstream consumes lower+diag only):
// 36 blocks instead of 64 -> ~44% less WMMA work / L2 traffic / atomics.
// LDS tiles stored column-major (col c, row/K r) so both A (needs transpose)
// and B fragments are contiguous 16B LDS loads per the CDNA5 16-bit layouts:
//   A 16x32: lane l holds M=l&15; K runs [s..s+7] and [16+s..16+s+7], s=(l&16)?8:0
//   B 32x16: lane l holds N=l&15; K runs [s..s+7] and [16+s..16+s+7], s=(l&16)?8:0
// ---------------------------------------------------------------------------
#define LSTR 40  // halfwords per LDS column: >=32, keeps 16B alignment, spreads banks

__device__ __forceinline__ v16bf lds_fragA(const __bf16* S, int colbase, int lane) {
  const int c = colbase + (lane & 15);
  const int s = (lane & 16) ? 8 : 0;
  v8bf lo = *(const v8bf*)&S[c * LSTR + s];
  v8bf hi = *(const v8bf*)&S[c * LSTR + 16 + s];
  return cat16(lo, hi);
}

__device__ __forceinline__ v16bf lds_fragB(const __bf16* S, int colbase, int lane) {
  const int c = colbase + (lane & 15);
  const int r = (lane & 16) ? 8 : 0;
  v8bf lo = *(const v8bf*)&S[c * LSTR + r];
  v8bf hi = *(const v8bf*)&S[c * LSTR + r + 16];
  return cat16(lo, hi);
}

__global__ __launch_bounds__(256)
void gram_kernel(const float* __restrict__ x, float* __restrict__ G,
                 int rows_per_chunk) {
  __shared__ __attribute__((aligned(16))) __bf16 As[64 * LSTR];
  __shared__ __attribute__((aligned(16))) __bf16 Bs[64 * LSTR];

  // flat lower-triangle block id -> (bi, bj), bi >= bj, bid = bi*(bi+1)/2 + bj
  const int bid = blockIdx.x;  // 0..35
  int bi = 0;
  while ((bi + 1) * (bi + 2) / 2 <= bid) ++bi;
  const int bj = bid - bi * (bi + 1) / 2;
  const bool diag = (bi == bj);

  const int i0 = bi * 64, j0 = bj * 64;  // G block origin
  const int tid = threadIdx.x;
  const int lane = tid & 31;
  const int w = tid >> 5;                // wave 0..7
  const int ti0 = (w >> 2) * 2;          // this wave's first row-tile (0 or 2)
  const int tj = w & 3;                  // this wave's col-tile

  const long rowbase = (long)blockIdx.y * rows_per_chunk;
  const int nsteps = rows_per_chunk >> 5;  // 32 rows of x per step

  // staging: thread -> row pair (2rp, 2rp+1) x col quad (c4 .. c4+3)
  const int rp = tid >> 4;
  const int c4 = (tid & 15) * 4;

  const __bf16* Bsrc = diag ? As : Bs;   // diagonal blocks: B reads the A tile

  v8f acc0 = {};
  v8f acc1 = {};

  for (int step = 0; step < nsteps; ++step) {
    const long r = rowbase + (long)step * 32 + 2 * rp;
    __syncthreads();  // protect previous iteration's fragment reads
    {
      // stage Xi block (cols i0..i0+63) -> As; Xj block -> Bs (bf16, col-major)
      const float4 ia = *(const float4*)(x + r * UNITS + i0 + c4);
      const float4 ib = *(const float4*)(x + (r + 1) * UNITS + i0 + c4);
      float ia4[4] = {ia.x, ia.y, ia.z, ia.w};
      float ib4[4] = {ib.x, ib.y, ib.z, ib.w};
#pragma unroll
      for (int i = 0; i < 4; ++i) {
        v2bf pi;
        pi.x = (__bf16)ia4[i]; pi.y = (__bf16)ib4[i];
        *(v2bf*)&As[(c4 + i) * LSTR + 2 * rp] = pi;
      }
      if (!diag) {
        const float4 ja = *(const float4*)(x + r * UNITS + j0 + c4);
        const float4 jb = *(const float4*)(x + (r + 1) * UNITS + j0 + c4);
        float ja4[4] = {ja.x, ja.y, ja.z, ja.w};
        float jb4[4] = {jb.x, jb.y, jb.z, jb.w};
#pragma unroll
        for (int i = 0; i < 4; ++i) {
          v2bf pj;
          pj.x = (__bf16)ja4[i]; pj.y = (__bf16)jb4[i];
          *(v2bf*)&Bs[(c4 + i) * LSTR + 2 * rp] = pj;
        }
      }
    }
    __syncthreads();

    const v16bf b = lds_fragB(Bsrc, tj * 16, lane);
    const v16bf a0 = lds_fragA(As, ti0 * 16, lane);
    const v16bf a1 = lds_fragA(As, (ti0 + 1) * 16, lane);
    acc0 = __builtin_amdgcn_wmma_f32_16x16x32_bf16(false, a0, false, b,
                                                   (short)0, acc0, false, false);
    acc1 = __builtin_amdgcn_wmma_f32_16x16x32_bf16(false, a1, false, b,
                                                   (short)0, acc1, false, false);
  }

  // reduce partial 16x16 tiles into G with f32 atomics
  const int colg = j0 + tj * 16 + (lane & 15);
  const int row0 = i0 + ti0 * 16 + ((lane & 16) ? 8 : 0);
#pragma unroll
  for (int v = 0; v < 8; ++v) {
    unsafeAtomicAdd(&G[(size_t)(row0 + v) * UNITS + colg], acc0[v]);
    unsafeAtomicAdd(&G[(size_t)(row0 + 16 + v) * UNITS + colg], acc1[v]);
  }
}

// ---------------------------------------------------------------------------
// Blocked Cholesky, panel factorization (one workgroup, panel lives in LDS)
// Factors columns [jb*64, jb*64+64) over rows [jb*64, 512).
// Only the lower triangle of G is ever consumed downstream.
// ---------------------------------------------------------------------------
__global__ __launch_bounds__(256)
void chol_panel_kernel(float* __restrict__ G, int jb) {
  extern __shared__ float Pnl[];  // R x 64, R = 512 - jb*64  (<=128KB of LDS)
  const int p0 = jb * 64;
  const int R = UNITS - p0;
  const int tid = threadIdx.x;

  for (int idx = tid; idx < R * 64; idx += 256) {
    const int r = idx >> 6, c = idx & 63;
    Pnl[idx] = G[(size_t)(p0 + r) * UNITS + p0 + c];
  }
  __syncthreads();

  for (int kk = 0; kk < 64; ++kk) {
    if (tid == 0) Pnl[kk * 64 + kk] = sqrtf(Pnl[kk * 64 + kk]);
    __syncthreads();
    const float dinv = 1.0f / Pnl[kk * 64 + kk];
    for (int r = kk + 1 + tid; r < R; r += 256) Pnl[r * 64 + kk] *= dinv;
    __syncthreads();
    const int c = kk + 1 + (tid & 63);
    if (c < 64) {
      const float lck = Pnl[c * 64 + kk];
      for (int r = c + (tid >> 6); r < R; r += 4)
        Pnl[r * 64 + c] -= Pnl[r * 64 + kk] * lck;
    }
    __syncthreads();
  }

  for (int idx = tid; idx < R * 64; idx += 256) {
    const int r = idx >> 6, c = idx & 63;
    G[(size_t)(p0 + r) * UNITS + p0 + c] = Pnl[idx];
  }
}

// ---------------------------------------------------------------------------
// Blocked Cholesky, trailing SYRK update: A22 -= L21 * L21^T (lower tiles)
// ---------------------------------------------------------------------------
__global__ __launch_bounds__(256)
void chol_update_kernel(float* __restrict__ G, int jb) {
  const int ti = blockIdx.y, tj = blockIdx.x;
  if (tj > ti) return;
  const int p0 = jb * 64;
  const int t0 = p0 + 64;
  const int i0 = t0 + ti * 64, j0 = t0 + tj * 64;
  __shared__ float Ai[64 * 65];
  __shared__ float Bj[64 * 65];
  const int tid = threadIdx.x;

  for (int idx = tid; idx < 64 * 64; idx += 256) {
    const int r = idx >> 6, k = idx & 63;
    Ai[r * 65 + k] = G[(size_t)(i0 + r) * UNITS + p0 + k];
    Bj[r * 65 + k] = G[(size_t)(j0 + r) * UNITS + p0 + k];
  }
  __syncthreads();

  const int c = tid & 63, rr = tid >> 6;
  float acc[16];
#pragma unroll
  for (int q = 0; q < 16; ++q) acc[q] = 0.0f;
  for (int k = 0; k < 64; ++k) {
    const float bv = Bj[c * 65 + k];
#pragma unroll
    for (int q = 0; q < 16; ++q) acc[q] += Ai[(rr + 4 * q) * 65 + k] * bv;
  }
#pragma unroll
  for (int q = 0; q < 16; ++q) {
    const int r = rr + 4 * q;
    G[(size_t)(i0 + r) * UNITS + j0 + c] -= acc[q];
  }
}

// ---------------------------------------------------------------------------
// Triangular inverse: one wave32 per column j of inv(L); forward substitution,
// lockstep within the wave (no barriers), __shfl broadcast of each pivot.
// Writes Wt = sqrt(512) * inv(L) as bf16 (Wt[a][b] nonzero for a >= b), which
// is exactly the B-operand layout the output GEMM wants.
// ---------------------------------------------------------------------------
__global__ __launch_bounds__(256)
void trinv_kernel(const float* __restrict__ G, __bf16* __restrict__ Wt) {
  const int lane = threadIdx.x & 31;
  const int j = blockIdx.x * 8 + (threadIdx.x >> 5);  // column 0..511

  for (int i = lane; i < j; i += 32) Wt[(size_t)i * UNITS + j] = (__bf16)0.0f;

  float acc[16];  // lane owns rows lane, lane+32, ..., lane+480
#pragma unroll
  for (int q = 0; q < 16; ++q) acc[q] = (lane + (q << 5) == j) ? 1.0f : 0.0f;

  for (int i = j; i < UNITS; ++i) {
    const int q = i >> 5, owner = i & 31;
    float av = 0.0f;
#pragma unroll
    for (int qq = 0; qq < 16; ++qq) av = (qq == q) ? acc[qq] : av;
    const float num = __shfl(av, owner, 32);
    const float m = num / G[(size_t)i * UNITS + i];
    if (lane == owner) Wt[(size_t)i * UNITS + j] = (__bf16)(m * SQRT_UNITS);
#pragma unroll
    for (int qq = 0; qq < 16; ++qq) {
      const int row = lane + (qq << 5);
      if (row > i) acc[qq] -= G[(size_t)row * UNITS + i] * m;
    }
  }
}

// ---------------------------------------------------------------------------
// Output GEMM: Y = x @ W  (W^T stored as Wt, bf16, L2-resident 512KB).
// Each workgroup: 32 rows x all 512 cols (x read from HBM exactly once).
// Wave w: rows r0 + (w&1)*16, col range (w>>1)*128 .. +127 (8 WMMA tiles).
// A built on the fly from f32 x: per lane two 8-float runs (K = s..s+7 and
// 16+s..16+s+7) -> matches the bf16 A-fragment layout after conversion.
// ---------------------------------------------------------------------------
__global__ __launch_bounds__(256)
void gemm_out_kernel(const float* __restrict__ x, const __bf16* __restrict__ Wt,
                     float* __restrict__ Y) {
  const int tid = threadIdx.x, lane = tid & 31, w = tid >> 5;
  const long r0 = (long)blockIdx.x * 32 + (w & 1) * 16;
  const int n0b = (w >> 1) * 128;
  const int m = lane & 15;
  const int s = (lane & 16) ? 8 : 0;

  v8f acc[8];
#pragma unroll
  for (int t = 0; t < 8; ++t) acc[t] = (v8f){};

  const float* xrow = x + (r0 + m) * UNITS;

  for (int k0 = 0; k0 < UNITS; k0 += 32) {
    const float* p = xrow + k0 + s;
    const float4 t0 = *(const float4*)(p);
    const float4 t1 = *(const float4*)(p + 4);
    const float4 t2 = *(const float4*)(p + 16);
    const float4 t3 = *(const float4*)(p + 20);
    v16bf a;
    a[0] = (__bf16)t0.x;  a[1] = (__bf16)t0.y;
    a[2] = (__bf16)t0.z;  a[3] = (__bf16)t0.w;
    a[4] = (__bf16)t1.x;  a[5] = (__bf16)t1.y;
    a[6] = (__bf16)t1.z;  a[7] = (__bf16)t1.w;
    a[8] = (__bf16)t2.x;  a[9] = (__bf16)t2.y;
    a[10] = (__bf16)t2.z; a[11] = (__bf16)t2.w;
    a[12] = (__bf16)t3.x; a[13] = (__bf16)t3.y;
    a[14] = (__bf16)t3.z; a[15] = (__bf16)t3.w;

#pragma unroll
    for (int t = 0; t < 8; ++t) {
      const __bf16* q = Wt + (size_t)(n0b + t * 16 + m) * UNITS + k0 + s;
      v8bf lo = *(const v8bf*)q;
      v8bf hi = *(const v8bf*)(q + 16);
      v16bf b = cat16(lo, hi);
      acc[t] = __builtin_amdgcn_wmma_f32_16x16x32_bf16(false, a, false, b,
                                                       (short)0, acc[t], false, false);
    }
  }

#pragma unroll
  for (int t = 0; t < 8; ++t) {
    const int col = n0b + t * 16 + m;
    const long rowb = r0 + ((lane & 16) ? 8 : 0);
#pragma unroll
    for (int v = 0; v < 8; ++v) Y[(rowb + v) * UNITS + col] = acc[t][v];
  }
}

// ---------------------------------------------------------------------------
extern "C" void kernel_launch(void* const* d_in, const int* in_sizes, int n_in,
                              void* d_out, int out_size, void* d_ws, size_t ws_size,
                              hipStream_t stream) {
  (void)n_in; (void)out_size; (void)ws_size;
  const float* x = (const float*)d_in[0];
  const int N = in_sizes[0] / UNITS;  // 262144

  float* G = (float*)d_ws;                                        // 1 MB f32
  __bf16* Wt = (__bf16*)((char*)d_ws + UNITS * UNITS * sizeof(float));  // 512 KB bf16
  float* Y = (float*)d_out;

  init_g_kernel<<<(UNITS * UNITS + 255) / 256, 256, 0, stream>>>(G);

  const int SPLITK = 32;
  const int rows_per_chunk = N / SPLITK;  // 8192
  gram_kernel<<<dim3(36, SPLITK), 256, 0, stream>>>(x, G, rows_per_chunk);

  for (int jb = 0; jb < 8; ++jb) {
    const int R = UNITS - jb * 64;
    chol_panel_kernel<<<1, 256, (size_t)R * 64 * sizeof(float), stream>>>(G, jb);
    const int nt = 7 - jb;
    if (nt > 0) chol_update_kernel<<<dim3(nt, nt), 256, 0, stream>>>(G, jb);
  }

  trinv_kernel<<<UNITS / 8, 256, 0, stream>>>(G, Wt);

  gemm_out_kernel<<<N / 32, 256, 0, stream>>>(x, Wt, Y);
}
